// Agent_77575699300542
// MI455X (gfx1250) — matrix operand order repaired and checked by reading, not compiled, not run on hardware
//
#include <hip/hip_runtime.h>

#define Tdim 256
#define Bdim 512
#define OBSD 512
#define ACTD 16
#define Hdim 128
#define G4H  512   // 4*H
#define LN_EPS 1e-5f

typedef __attribute__((ext_vector_type(16))) _Float16 v16h;
typedef __attribute__((ext_vector_type(8)))  _Float16 v8h;
typedef __attribute__((ext_vector_type(4)))  _Float16 v4h;
typedef __attribute__((ext_vector_type(8)))  float    v8f;
typedef __attribute__((ext_vector_type(4)))  unsigned v4u;
typedef __attribute__((ext_vector_type(8)))  unsigned v8u;

__device__ __forceinline__ v8f wmma_f16(v16h a, v16h b, v8f c) {
  // D = A(16x32 f16) * B(32x16 f16) + C(16x16 f32)
  return __builtin_amdgcn_wmma_f32_16x16x32_f16(false, a, false, b, (short)0, c, false, false);
}

__device__ __forceinline__ float sigm(float x)  { return 1.f / (1.f + __expf(-x)); }
__device__ __forceinline__ float ftanh(float x) { return 2.f / (1.f + __expf(-2.f * x)) - 1.f; }

// A fragment (16x32 f16) from row-major source (lda elements).
// Per ISA 7.12.2: two contiguous 8-half runs per lane at k = hf*8 (+16).
__device__ __forceinline__ v16h frag_a16(const _Float16* p, int lda) {
  const int lane = threadIdx.x & 31;
  const int m = lane & 15, hf = lane >> 4;
  const _Float16* q = p + m * lda + hf * 8;
  const v8h lo = *(const v8h*)q;
  const v8h hi = *(const v8h*)(q + 16);
  v16h a;
#pragma unroll
  for (int i = 0; i < 8; ++i) { a[i] = lo[i]; a[8 + i] = hi[i]; }
  return a;
}

// B fragment (32x16) = W^T, W row-major [N][K]; two contiguous 8-half runs
// per lane at k = hf*8 (+16), n = lane%16.
__device__ __forceinline__ v16h frag_bT16(const _Float16* p, int ldw) {
  const int lane = threadIdx.x & 31;
  const int n = lane & 15, hf = lane >> 4;
  const _Float16* q = p + n * ldw + hf * 8;
  const v8h lo = *(const v8h*)q;
  const v8h hi = *(const v8h*)(q + 16);
  v16h b;
#pragma unroll
  for (int i = 0; i < 8; ++i) { b[i] = lo[i]; b[8 + i] = hi[i]; }
  return b;
}

// ---------------------------------------------------------------------------
// fp32 -> f16 weight conversion (one-shot)
// ---------------------------------------------------------------------------
__global__ void __launch_bounds__(256)
cvt_f16_kernel(const float* __restrict__ in, _Float16* __restrict__ out, int n) {
  const int i = (blockIdx.x * 256 + threadIdx.x) * 4;
  if (i < n) {
    const float4 f = *(const float4*)(in + i);
    v4h o;
    o[0] = (_Float16)f.x; o[1] = (_Float16)f.y;
    o[2] = (_Float16)f.z; o[3] = (_Float16)f.w;
    *(v4h*)(out + i) = o;
  }
}

// ---------------------------------------------------------------------------
// Fused feature MLP + LSTM input projection.
// gx is stored in WMMA-fragment order: [tile16][wave(j/16)][gate][lane][v].
// ---------------------------------------------------------------------------
__global__ void __launch_bounds__(256)
feat_proj_kernel(const float* __restrict__ x,
                 const _Float16* __restrict__ W1h, const float* __restrict__ b1,
                 const _Float16* __restrict__ W2h, const float* __restrict__ b2,
                 const float* __restrict__ lg, const float* __restrict__ lb,
                 const _Float16* __restrict__ Wihh,
                 const float* __restrict__ bih, const float* __restrict__ bhh,
                 _Float16* __restrict__ gx) {
  __shared__ _Float16 xs[16 * OBSD];   // 16 KB
  __shared__ _Float16 h1s[16 * 256];   //  8 KB
  __shared__ float    h2s[16 * Hdim];  //  8 KB
  __shared__ _Float16 fs[16 * Hdim];   //  4 KB
  __shared__ float    bsum[G4H];       //  2 KB
  __shared__ float    b1s[256], b2s[128], lgs[128], lbs[128];

  const int tid  = threadIdx.x;
  const int lane = tid & 31, wv = tid >> 5;
  const int hf   = lane >> 4, nn = lane & 15;
  const int row0 = blockIdx.x * 16;

  for (int i = tid * 4; i < 16 * OBSD; i += 1024) {
    const float4 f = *(const float4*)&x[(size_t)(row0 + (i >> 9)) * OBSD + (i & (OBSD - 1))];
    xs[i] = (_Float16)f.x; xs[i + 1] = (_Float16)f.y;
    xs[i + 2] = (_Float16)f.z; xs[i + 3] = (_Float16)f.w;
  }
  for (int i = tid; i < G4H; i += 256) bsum[i] = bih[i] + bhh[i];
  if (tid < 256) b1s[tid] = b1[tid];
  if (tid < 128) { b2s[tid] = b2[tid]; lgs[tid] = lg[tid]; lbs[tid] = lb[tid]; }
  __syncthreads();

  // GEMM1: wave wv -> h1 columns [32*wv, 32*wv+32)
  {
    v8f acc0 = {}, acc1 = {};
    const int nc0 = wv * 32, nc1 = wv * 32 + 16;
#pragma unroll 4
    for (int kt = 0; kt < 16; ++kt) {
      v16h a = frag_a16(&xs[kt * 32], OBSD);
      acc0 = wmma_f16(a, frag_bT16(&W1h[(size_t)nc0 * OBSD + kt * 32], OBSD), acc0);
      acc1 = wmma_f16(a, frag_bT16(&W1h[(size_t)nc1 * OBSD + kt * 32], OBSD), acc1);
    }
#pragma unroll
    for (int v = 0; v < 8; ++v) {
      const int m = v + hf * 8;
      float t0 = acc0[v] + b1s[nc0 + nn]; t0 = t0 > 0.f ? t0 : 0.f;
      float t1 = acc1[v] + b1s[nc1 + nn]; t1 = t1 > 0.f ? t1 : 0.f;
      h1s[m * 256 + nc0 + nn] = (_Float16)t0;
      h1s[m * 256 + nc1 + nn] = (_Float16)t1;
    }
  }
  __syncthreads();

  // GEMM2: wave wv -> h2 columns [16*wv, 16*wv+16)
  {
    v8f acc = {};
    const int nc = wv * 16;
#pragma unroll 2
    for (int kt = 0; kt < 8; ++kt)
      acc = wmma_f16(frag_a16(&h1s[kt * 32], 256),
                     frag_bT16(&W2h[(size_t)nc * 256 + kt * 32], 256), acc);
#pragma unroll
    for (int v = 0; v < 8; ++v)
      h2s[(v + hf * 8) * Hdim + nc + nn] = acc[v] + b2s[nc + nn];
  }
  __syncthreads();

  // layernorm + relu -> fs ; 16 threads per row, shfl_xor butterfly reduction
  {
    const int m = tid >> 4, q = tid & 15;
    float s = 0.f, s2 = 0.f, vals[8];
#pragma unroll
    for (int k = 0; k < 8; ++k) {
      const float v = h2s[m * Hdim + q * 8 + k];
      vals[k] = v; s += v; s2 += v * v;
    }
#pragma unroll
    for (int d = 1; d < 16; d <<= 1) {
      s += __shfl_xor(s, d, 16);
      s2 += __shfl_xor(s2, d, 16);
    }
    const float mu = s * (1.f / Hdim);
    const float rstd = rsqrtf(s2 * (1.f / Hdim) - mu * mu + LN_EPS);
    v8h o;
#pragma unroll
    for (int k = 0; k < 8; ++k) {
      const int j = q * 8 + k;
      const float v = (vals[k] - mu) * rstd * lgs[j] + lbs[j];
      o[k] = (_Float16)(v > 0.f ? v : 0.f);
    }
    *(v8h*)&fs[m * Hdim + q * 8] = o;
  }
  __syncthreads();

  // GEMM3: gx = f @ Wih^T + (bih+bhh); stored fragment-major per 16-row tile.
#pragma unroll
  for (int u = 0; u < 4; ++u) {
    const int nc = wv * 64 + u * 16;
    const int g = nc >> 7;            // gate index (i,f,g,o)
    const int wvL = (nc >> 4) & 7;    // consuming wave in lstm kernel
    v8f acc = {};
#pragma unroll
    for (int kt = 0; kt < 4; ++kt)
      acc = wmma_f16(frag_a16(&fs[kt * 32], Hdim),
                     frag_bT16(&Wihh[(size_t)nc * Hdim + kt * 32], Hdim), acc);
    v8h st;
#pragma unroll
    for (int v = 0; v < 8; ++v)
      st[v] = (_Float16)(acc[v] + bsum[nc + nn]);
    _Float16* dst = gx + ((((size_t)blockIdx.x * 8 + wvL) * 4 + g) * 256) + lane * 8;
    *(v8h*)dst = st;
  }
}

// ---------------------------------------------------------------------------
// LSTM scan + head. One workgroup per 16 batch rows, full T=256 scan.
// Whh (f16, 128 KB) staged into LDS by the Tensor Data Mover (wave 0);
// head weights staged with global_load_async_to_lds_b128.
// ---------------------------------------------------------------------------
template <int ACTDIM>
__global__ void __launch_bounds__(256)
lstm_head_kernel(const _Float16* __restrict__ gx,
                 const int* __restrict__ done,
                 const float* __restrict__ h0, const float* __restrict__ c0,
                 const _Float16* __restrict__ Whh16,
                 const float* __restrict__ lg, const float* __restrict__ lb,
                 const _Float16* __restrict__ Wh16, const float* __restrict__ bh,
                 const int* __restrict__ action,
                 float* __restrict__ out0,   // actor: log_prob ; critic: value
                 float* __restrict__ out1,   // actor: entropy  ; critic: unused
                 float* __restrict__ outH, float* __restrict__ outC) {
  __shared__ _Float16 whh_s[G4H * Hdim];    // 128 KB
  __shared__ _Float16 h_s[16 * Hdim];       //   4 KB
  __shared__ _Float16 ln_s[16 * Hdim];      //   4 KB
  __shared__ _Float16 wh_s[ACTD * Hdim];    //   4 KB
  __shared__ float lgs[Hdim], lbs[Hdim];
  __shared__ float bhs[ACTD];
  __shared__ float dmask[16];
  __shared__ float logit_s[16 * ACTD];

  const int tid  = threadIdx.x;
  const int lane = tid & 31, wv = tid >> 5;
  const int hf   = lane >> 4, nn = lane & 15;
  const int r0   = blockIdx.x * 16;
  const int jb   = wv * 16;   // this wave's hidden-unit block

  // ---- TDM: DMA Whh (f16, 131072 B = 32768 dwords) into LDS, wave 0 only ----
  if (wv == 0) {
    const unsigned lds0 = (unsigned)(size_t)&whh_s[0];
    const unsigned long long ga = (unsigned long long)(size_t)Whh16;
    v4u g0;
    g0[0] = 1u;                                    // count=1 (valid user D#)
    g0[1] = lds0;                                  // lds_addr (bytes)
    g0[2] = (unsigned)ga;                          // global_addr[31:0]
    g0[3] = (unsigned)(ga >> 32) | 0x80000000u;    // global_addr[56:32] | type=2
    v8u g1;
    g1[0] = 0x00020000u;   // workgroup_mask=0, data_size=2 (4-byte units)
    g1[1] = 0x80000000u;   // tensor_dim0 = 32768 (bits[79:48], low half)
    g1[2] = 0x00010000u;   // tensor_dim0 hi = 0 ; tensor_dim1 = 1
    g1[3] = 0x80000000u;   // tile_dim0 = 32768 (bits[127:112])
    g1[4] = 0u;            // tile_dim1 = 0 (unused), tile_dim2 = 0
    g1[5] = 32768u;        // tensor_dim0_stride lo
    g1[6] = 0u;
    g1[7] = 0u;
    asm volatile("tensor_load_to_lds %0, %1" :: "s"(g0), "s"(g1) : "memory");
    __builtin_amdgcn_s_wait_tensorcnt(0);
  }

  // ---- async copy of head weights into LDS (16 B per lane per issue) ----
  {
    const unsigned lds0 = (unsigned)(size_t)&wh_s[0];
    for (int i = tid; i < (ACTDIM * Hdim) / 8; i += 256) {
      const unsigned laddr = lds0 + (unsigned)(i * 16);
      const unsigned long long ga = (unsigned long long)(size_t)(Wh16 + i * 8);
      asm volatile("global_load_async_to_lds_b128 %0, %1, off"
                   :: "v"(laddr), "v"(ga) : "memory");
    }
  }

  for (int i = tid; i < 16 * Hdim; i += 256)
    h_s[i] = (_Float16)h0[(size_t)(r0 + (i >> 7)) * Hdim + (i & 127)];
  if (tid < Hdim) { lgs[tid] = lg[tid]; lbs[tid] = lb[tid]; }
  if (tid < ACTDIM) bhs[tid] = bh[tid];

  float c[8];
#pragma unroll
  for (int v = 0; v < 8; ++v)
    c[v] = c0[(size_t)(r0 + v + hf * 8) * Hdim + jb + nn];

  asm volatile("s_wait_asynccnt 0" ::: "memory");
  __syncthreads();

  // Hoist Whh^T fragments: gate g, k-step kt -> columns g*128 + jb .. +16
  v16h bw[4][4];
#pragma unroll
  for (int g = 0; g < 4; ++g)
#pragma unroll
    for (int kt = 0; kt < 4; ++kt)
      bw[g][kt] = frag_bT16(&whh_s[(size_t)(g * 128 + jb) * Hdim + kt * 32], Hdim);

  const int m16 = tid >> 4, q16 = tid & 15;   // 16-thread row groups

  for (int t = 0; t < Tdim; ++t) {
    if (tid < 16) dmask[tid] = 1.f - (float)done[t * Bdim + r0 + tid];
    if (t + 1 < Tdim)   // prefetch next timestep's 16 KB gx tile
      __builtin_prefetch(gx + ((size_t)((t + 1) * (Bdim / 16) + blockIdx.x) * 8192)
                            + tid * 32, 0, 1);
    __syncthreads();

    {   // reset h where done (vectorized, one v8h chunk per thread)
      const float dm = dmask[m16];
      v8h hv = *(const v8h*)&h_s[m16 * Hdim + q16 * 8];
#pragma unroll
      for (int k = 0; k < 8; ++k) hv[k] = (_Float16)((float)hv[k] * dm);
      *(v8h*)&h_s[m16 * Hdim + q16 * 8] = hv;
    }
#pragma unroll
    for (int v = 0; v < 8; ++v) c[v] *= dmask[v + hf * 8];
    __syncthreads();

    v16h ah[4];
#pragma unroll
    for (int kt = 0; kt < 4; ++kt) ah[kt] = frag_a16(&h_s[kt * 32], Hdim);

    // gx in fragment order: [(t*32+blk)*8 + wv][g][lane][v]
    const _Float16* gxt = gx
        + (((size_t)(t * (Bdim / 16) + blockIdx.x) * 8 + wv) * 4) * 256 + lane * 8;
    v8f gacc[4];
#pragma unroll
    for (int g = 0; g < 4; ++g) {
      const v8h gv = *(const v8h*)(gxt + g * 256);
      v8f acc;
#pragma unroll
      for (int v = 0; v < 8; ++v) acc[v] = (float)gv[v];
#pragma unroll
      for (int kt = 0; kt < 4; ++kt) acc = wmma_f16(ah[kt], bw[g][kt], acc);
      gacc[g] = acc;
    }

    // gate order: i, f, g, o
#pragma unroll
    for (int v = 0; v < 8; ++v) {
      const float cn = sigm(gacc[1][v]) * c[v] + sigm(gacc[0][v]) * ftanh(gacc[2][v]);
      c[v] = cn;
      h_s[(v + hf * 8) * Hdim + jb + nn] = (_Float16)(sigm(gacc[3][v]) * ftanh(cn));
    }
    __syncthreads();

    // layernorm of updated h -> ln_s (16 threads/row, butterfly reduce)
    {
      const v8h hv = *(const v8h*)&h_s[m16 * Hdim + q16 * 8];
      float s = 0.f, s2 = 0.f, vals[8];
#pragma unroll
      for (int k = 0; k < 8; ++k) {
        const float v = (float)hv[k];
        vals[k] = v; s += v; s2 += v * v;
      }
#pragma unroll
      for (int d = 1; d < 16; d <<= 1) {
        s += __shfl_xor(s, d, 16);
        s2 += __shfl_xor(s2, d, 16);
      }
      const float mu = s * (1.f / Hdim);
      const float rstd = rsqrtf(s2 * (1.f / Hdim) - mu * mu + LN_EPS);
      v8h lnv;
#pragma unroll
      for (int k = 0; k < 8; ++k) {
        const int j = q16 * 8 + k;
        lnv[k] = (_Float16)((vals[k] - mu) * rstd * lgs[j] + lbs[j]);
      }
      *(v8h*)&ln_s[m16 * Hdim + q16 * 8] = lnv;
    }
    __syncthreads();

    if (ACTDIM == ACTD) {
      if (wv == 0) {  // logits = ln @ Wh^T + bh
        v8f acc = {};
#pragma unroll
        for (int kt = 0; kt < 4; ++kt)
          acc = wmma_f16(frag_a16(&ln_s[kt * 32], Hdim),
                         frag_bT16(&wh_s[kt * 32], Hdim), acc);
#pragma unroll
        for (int v = 0; v < 8; ++v)
          logit_s[(v + hf * 8) * ACTD + nn] = acc[v] + bhs[nn];
      }
      __syncthreads();
      {  // log-softmax: thread = (row m16, action q16)
        const float l = logit_s[m16 * ACTD + q16];
        float mx = l;
#pragma unroll
        for (int d = 1; d < 16; d <<= 1) mx = fmaxf(mx, __shfl_xor(mx, d, 16));
        float se = __expf(l - mx);
#pragma unroll
        for (int d = 1; d < 16; d <<= 1) se += __shfl_xor(se, d, 16);
        const float lse = mx + __logf(se);
        const float lp = l - lse;
        float ent = -__expf(lp) * lp;
#pragma unroll
        for (int d = 1; d < 16; d <<= 1) ent += __shfl_xor(ent, d, 16);
        const int row = t * Bdim + r0 + m16;
        if (q16 == 0) out1[row] = ent;
        if (q16 == action[row]) out0[row] = lp;
      }
    } else {
      // value = ln . wh + bh ; 16 threads/row dot, butterfly reduce
      const v8h lv = *(const v8h*)&ln_s[m16 * Hdim + q16 * 8];
      const v8h wvv = *(const v8h*)&wh_s[q16 * 8];
      float s = 0.f;
#pragma unroll
      for (int k = 0; k < 8; ++k) s += (float)lv[k] * (float)wvv[k];
#pragma unroll
      for (int d = 1; d < 16; d <<= 1) s += __shfl_xor(s, d, 16);
      if (q16 == 0) out0[t * Bdim + r0 + m16] = s + bhs[0];
    }
  }

  for (int i = tid; i < 16 * Hdim; i += 256)
    outH[(size_t)(r0 + (i >> 7)) * Hdim + (i & 127)] = (float)h_s[i];
#pragma unroll
  for (int v = 0; v < 8; ++v)
    outC[(size_t)(r0 + v + hf * 8) * Hdim + jb + nn] = c[v];
}

__global__ void __launch_bounds__(256)
action_copy_kernel(const int* __restrict__ a, float* __restrict__ o, int n) {
  const int i = blockIdx.x * blockDim.x + threadIdx.x;
  if (i < n) o[i] = (float)a[i];
}

extern "C" void kernel_launch(void* const* d_in, const int* in_sizes, int n_in,
                              void* d_out, int out_size, void* d_ws, size_t ws_size,
                              hipStream_t stream) {
  (void)in_sizes; (void)n_in; (void)out_size; (void)ws_size;
  const float* x      = (const float*)d_in[0];
  const int*   done   = (const int*)d_in[1];
  const int*   action = (const int*)d_in[2];
  const float* a_h0   = (const float*)d_in[3];
  const float* a_c0   = (const float*)d_in[4];
  const float* c_h0   = (const float*)d_in[5];
  const float* c_c0   = (const float*)d_in[6];
  // per-branch params: W1,b1,W2,b2,ln1_g,ln1_b,Wih,Whh,bih,bhh,ln2_g,ln2_b,Wh,bh
  const float* A[14];
  const float* C[14];
  for (int i = 0; i < 14; ++i) {
    A[i] = (const float*)d_in[7 + i];
    C[i] = (const float*)d_in[21 + i];
  }

  float* out = (float*)d_out;
  const int TB = Tdim * Bdim;
  float* o_act = out;
  float* o_lp  = out + (size_t)TB;
  float* o_ent = out + (size_t)2 * TB;
  float* o_val = out + (size_t)3 * TB;
  float* o_ahT = out + (size_t)4 * TB;
  float* o_acT = o_ahT + (size_t)Bdim * Hdim;
  float* o_chT = o_acT + (size_t)Bdim * Hdim;
  float* o_ccT = o_chT + (size_t)Bdim * Hdim;

  // workspace layout (f16 elements)
  _Float16* gx_a = (_Float16*)d_ws;
  _Float16* gx_c = gx_a + (size_t)TB * G4H;
  _Float16* wb   = gx_c + (size_t)TB * G4H;
  const size_t nW1 = 256 * OBSD, nW2 = Hdim * 256, nWih = G4H * Hdim,
               nWhh = G4H * Hdim, nWh = ACTD * Hdim;
  _Float16* A16[5];  // W1, W2, Wih, Whh, Wh
  _Float16* C16[5];
  {
    _Float16* p = wb;
    A16[0] = p; p += nW1;  A16[1] = p; p += nW2;  A16[2] = p; p += nWih;
    A16[3] = p; p += nWhh; A16[4] = p; p += nWh;
    C16[0] = p; p += nW1;  C16[1] = p; p += nW2;  C16[2] = p; p += nWih;
    C16[3] = p; p += nWhh; C16[4] = p; p += nWh;
  }

  const dim3 blk(256);
  const int srcIdx[5] = {0, 2, 6, 7, 12};  // W1, W2, Wih, Whh, Wh
  const int cnt[5] = {(int)nW1, (int)nW2, (int)nWih, (int)nWhh, (int)nWh};
  for (int i = 0; i < 5; ++i) {
    cvt_f16_kernel<<<(cnt[i] / 4 + 255) / 256, blk, 0, stream>>>(A[srcIdx[i]], A16[i], cnt[i]);
    const int cc = (i == 4) ? Hdim : cnt[i];  // critic head is [1,128]
    cvt_f16_kernel<<<(cc / 4 + 255) / 256, blk, 0, stream>>>(C[srcIdx[i]], C16[i], cc);
  }

  feat_proj_kernel<<<TB / 16, blk, 0, stream>>>(
      x, A16[0], A[1], A16[1], A[3], A[4], A[5], A16[2], A[8], A[9], gx_a);
  feat_proj_kernel<<<TB / 16, blk, 0, stream>>>(
      x, C16[0], C[1], C16[1], C[3], C[4], C[5], C16[2], C[8], C[9], gx_c);
  lstm_head_kernel<ACTD><<<Bdim / 16, blk, 0, stream>>>(
      gx_a, done, a_h0, a_c0, A16[3], A[10], A[11], A16[4], A[13], action,
      o_lp, o_ent, o_ahT, o_acT);
  lstm_head_kernel<1><<<Bdim / 16, blk, 0, stream>>>(
      gx_c, done, c_h0, c_c0, C16[3], C[10], C[11], C16[4], C[13], action,
      o_val, nullptr, o_chT, o_ccT);
  action_copy_kernel<<<(TB + 255) / 256, blk, 0, stream>>>(action, o_act, TB);
}
